// GCAModel_49409303773853
// MI455X (gfx1250) — matrix operand order repaired and checked by reading, not compiled, not run on hardware
//
#include <hip/hip_runtime.h>
#include <hip/hip_bf16.h>

// ---------------------------------------------------------------------------
// GCAModel (NRMS-style) forward for MI455X / gfx1250, wave32 + WMMA f16.
// Shapes: VOCAB=50000 D=300 H=16 V=16 Q=200 | B=32 CDD=5 HIS=50 L=32 K=20
// ---------------------------------------------------------------------------

typedef __attribute__((ext_vector_type(16))) _Float16 v16h;
typedef __attribute__((ext_vector_type(8)))  float    v8f;

#define DPAD 320           // D=300 padded to 10 k-steps of 32
#define KT_Q 10            // k-tiles (K=320) for D-contraction GEMMs
#define NT_Q 20            // n-tiles for N=320 outputs
#define KT_K 8             // k-tiles for K=256 (H*V) contraction
#define NT_K 13            // n-tiles for N=208 (Q=200 padded)
#define NCAND 160          // B*CDD
#define NITEM 1760         // B*(CDD+HIS)

// ---------------- WMMA fragment helpers (A/B: contiguous k = 16*(lane>>4) + j) ----------------

__device__ __forceinline__ v8f wmma_f16(v16h a, v16h b, v8f c) {
  return __builtin_amdgcn_wmma_f32_16x16x32_f16(false, a, false, b, (short)0, c,
                                                false, false);
}

// A fragment 16x32 from row-major f16 LDS [.. x ld]
__device__ __forceinline__ v16h ldsA_f16(const _Float16* base, int mBase,
                                         int kBase, int ld, int lane) {
  int m  = mBase + (lane & 15);
  int kh = (lane >> 4) * 16;          // lanes 0-15: K 0-15; 16-31: K 16-31
  const _Float16* p = base + m * ld + kBase + kh;
  v16h a;
#pragma unroll
  for (int j = 0; j < 8; ++j) { a[j] = p[j]; a[8 + j] = p[8 + j]; }
  return a;
}

// B fragment 32x16 where B[k][n] = base[(nBase+n)*ld + kBase + k] (k contiguous)
__device__ __forceinline__ v16h ldsB_rows(const _Float16* base, int kBase,
                                          int nBase, int ld, int lane) {
  int n  = nBase + (lane & 15);
  int kh = (lane >> 4) * 16;          // lanes 0-15: K 0-15; 16-31: K 16-31
  const _Float16* p = base + n * ld + kBase + kh;
  v16h b;
#pragma unroll
  for (int j = 0; j < 16; ++j) b[j] = p[j];
  return b;
}

// B fragment 32x16 where B[k][n] = base[(kBase+k)*ld + nBase + n] (k strided)
__device__ __forceinline__ v16h ldsB_cols(const _Float16* base, int kBase,
                                          int nBase, int ld, int lane) {
  int n  = nBase + (lane & 15);
  int k0 = kBase + (lane >> 4) * 16;
  v16h b;
#pragma unroll
  for (int j = 0; j < 16; ++j) b[j] = base[(k0 + j) * ld + n];
  return b;
}

// C/D 16x16 f32 accumulator store: VGPR r -> row r (lanes 0-15) / r+8 (16-31)
__device__ __forceinline__ void stC_f16(_Float16* base, v8f c, int mBase,
                                        int nBase, int ld, int lane) {
  int n = nBase + (lane & 15);
  int m = mBase + ((lane >> 4) * 8);
#pragma unroll
  for (int r = 0; r < 8; ++r) base[(m + r) * ld + n] = (_Float16)c[r];
}

__device__ __forceinline__ void stC_f32(float* base, v8f c, int mBase, int nBase,
                                        int ld, int lane, float scale) {
  int n = nBase + (lane & 15);
  int m = mBase + ((lane >> 4) * 8);
#pragma unroll
  for (int r = 0; r < 8; ++r) base[(m + r) * ld + n] = c[r] * scale;
}

__device__ __forceinline__ float gumbel_hash(unsigned x) {
  x ^= x >> 16; x *= 0x7feb352dU; x ^= x >> 15; x *= 0x846ca68bU; x ^= x >> 16;
  float u = (float)(x >> 8) * (1.0f / 16777216.0f);
  u = fminf(fmaxf(u, 1e-7f), 1.0f - 1e-7f);
  return -__logf(-__logf(u));
}

// ---------------- prep kernels: repack weights into B-fragment layout -------
// tile = 32 lanes x 16 halfs (512 halfs); lane (n = lane&15, khalf = lane>>4),
// element j -> B[kt*32 + khalf*16 + j][nt*16 + n].

__global__ void prep_wq(const float* __restrict__ Wq, _Float16* __restrict__ wqt) {
  int idx = blockIdx.x * 256 + threadIdx.x;
  const int total = 16 * KT_Q * NT_Q * 512;
  if (idx >= total) return;
  int j = idx & 15, lane = (idx >> 4) & 31, tile = idx >> 9;
  int nt = tile % NT_Q, t2 = tile / NT_Q, kt = t2 % KT_Q, h = t2 / KT_Q;
  int d = kt * 32 + (lane >> 4) * 16 + j;   // contraction index (k)
  int e = nt * 16 + (lane & 15);            // output index (n)
  float v = (d < 300 && e < 300) ? Wq[((size_t)h * 300 + e) * 300 + d] : 0.f;
  wqt[idx] = (_Float16)v;                   // B[d][e] = Wq[h][e][d]
}

__global__ void prep_wv(const float* __restrict__ Wv, _Float16* __restrict__ wvt) {
  int idx = blockIdx.x * 256 + threadIdx.x;
  const int total = 16 * KT_Q * 512;
  if (idx >= total) return;
  int j = idx & 15, lane = (idx >> 4) & 31, tile = idx >> 9;
  int kt = tile % KT_Q, h = tile / KT_Q;
  int d = kt * 32 + (lane >> 4) * 16 + j;
  int vch = lane & 15;
  float v = (d < 300) ? Wv[((size_t)h * 16 + vch) * 300 + d] : 0.f;
  wvt[idx] = (_Float16)v;                   // B[d][v] = Wv[h][v][d]
}

__global__ void prep_wk(const float* __restrict__ Wk, _Float16* __restrict__ wkt) {
  int idx = blockIdx.x * 256 + threadIdx.x;
  const int total = KT_K * NT_K * 512;
  if (idx >= total) return;
  int j = idx & 15, lane = (idx >> 4) & 31, tile = idx >> 9;
  int nt = tile % NT_K, kt = tile / NT_K;
  int v = kt * 32 + (lane >> 4) * 16 + j;   // value-channel (k), < 256
  int q = nt * 16 + (lane & 15);
  float w = (q < 200) ? Wk[(size_t)q * 256 + v] : 0.f;
  wkt[idx] = (_Float16)w;                   // B[v][q] = Wk[q][v]
}

// ---------------- encode: per news item, full multi-head attn + pooling -----

__global__ void __launch_bounds__(256)
encode_kernel(const int* __restrict__ cand, const int* __restrict__ clicked,
              const float* __restrict__ emb, const float* __restrict__ qw,
              const float* __restrict__ bk, const _Float16* __restrict__ wqt,
              const _Float16* __restrict__ wvt, const _Float16* __restrict__ wkt,
              float* __restrict__ attnOut) {
  __shared__ __align__(32) _Float16 embF[32 * DPAD];   // token embeddings f16
  __shared__ __align__(32) _Float16 qbuf[32 * DPAD];   // Q, then ctx (reused)
  __shared__ __align__(32) _Float16 valb[32 * 256];    // concat head outputs
  __shared__ float sbuf[32 * 32];                      // attention logits
  __shared__ __align__(32) _Float16 abuf[32 * 32];     // softmaxed attention
  __shared__ float sArr[32], aArr[32], red[2];
  __shared__ int tokL[32];

  const int tid = threadIdx.x, wave = tid >> 5, lane = tid & 31;
  const int slot = blockIdx.x;
  const int* tok = (slot < NCAND) ? (cand + slot * 32)
                                  : (clicked + (size_t)(slot - NCAND) * 32);
  if (tid < 32) tokL[tid] = tok[tid];
  __syncthreads();
  for (int idx = tid; idx < 32 * DPAD; idx += 256) {
    int r = idx / DPAD, d = idx - r * DPAD;
    embF[idx] = (d < 300) ? (_Float16)emb[(size_t)tokL[r] * 300 + d]
                          : (_Float16)0.f;
  }
  __syncthreads();

  const float scale = 0.057735026918962584f;  // 1/sqrt(300)

  for (int h = 0; h < 16; ++h) {
    if (tid == 0)
      __builtin_prefetch(wqt + (size_t)((h + 1) & 15) * KT_Q * NT_Q * 512, 0, 1);

    // Q = emb @ Wq[h]^T  (32 x 320, K=320)
    for (int t = wave; t < 2 * NT_Q; t += 8) {
      int mt = t & 1, nt = t >> 1;
      v8f acc = {};
      const v16h* bp =
          (const v16h*)(wqt + ((size_t)(h * KT_Q) * NT_Q + nt) * 512);
#pragma unroll
      for (int kt = 0; kt < KT_Q; ++kt) {
        v16h a = ldsA_f16(embF, mt * 16, kt * 32, DPAD, lane);
        v16h b = bp[(size_t)kt * NT_Q * 32 + lane];
        acc = wmma_f16(a, b, acc);
      }
      stC_f16(qbuf, acc, mt * 16, nt * 16, DPAD, lane);
    }
    __syncthreads();

    // S = scale * Q @ emb^T  (32 x 32, K=320)
    if (wave < 4) {
      int mt = wave >> 1, nt = wave & 1;
      v8f acc = {};
#pragma unroll
      for (int kt = 0; kt < KT_Q; ++kt) {
        v16h a = ldsA_f16(qbuf, mt * 16, kt * 32, DPAD, lane);
        v16h b = ldsB_rows(embF, kt * 32, nt * 16, DPAD, lane);
        acc = wmma_f16(a, b, acc);
      }
      stC_f32(sbuf, acc, mt * 16, nt * 16, 32, lane, scale);
    }
    __syncthreads();

    // row softmax -> abuf
    if (tid < 32) {
      const float* row = sbuf + tid * 32;
      float mx = row[0];
      for (int j = 1; j < 32; ++j) mx = fmaxf(mx, row[j]);
      float sum = 0.f, e[32];
      for (int j = 0; j < 32; ++j) { e[j] = __expf(row[j] - mx); sum += e[j]; }
      float inv = 1.f / sum;
      for (int j = 0; j < 32; ++j) abuf[tid * 32 + j] = (_Float16)(e[j] * inv);
    }
    __syncthreads();

    // ctx = A @ emb  (32 x 320, K=32) -> overwrite qbuf
    for (int t = wave; t < 2 * NT_Q; t += 8) {
      int mt = t & 1, nt = t >> 1;
      v16h a = ldsA_f16(abuf, mt * 16, 0, 32, lane);
      v16h b = ldsB_cols(embF, 0, nt * 16, DPAD, lane);
      v8f acc = {};
      acc = wmma_f16(a, b, acc);
      stC_f16(qbuf, acc, mt * 16, nt * 16, DPAD, lane);
    }
    __syncthreads();

    // head_out = ctx @ Wv[h]^T  (32 x 16, K=320) -> valb[:, h*16 ..)
    if (wave < 2) {
      int mt = wave;
      v8f acc = {};
      const v16h* bp = (const v16h*)(wvt + (size_t)(h * KT_Q) * 512);
#pragma unroll
      for (int kt = 0; kt < KT_Q; ++kt) {
        v16h a = ldsA_f16(qbuf, mt * 16, kt * 32, DPAD, lane);
        v16h b = bp[(size_t)kt * 32 + lane];
        acc = wmma_f16(a, b, acc);
      }
      stC_f16(valb, acc, mt * 16, h * 16, 256, lane);
    }
    __syncthreads();
  }

  // keyt = tanh(val @ Wk^T + bk); s = scale * qw . keyt  (fused, no keyt store)
  if (tid < 32) sArr[tid] = 0.f;
  __syncthreads();
  for (int t = wave; t < 2 * NT_K; t += 8) {
    int mt = t & 1, nt = t >> 1;
    v8f acc = {};
    const v16h* bp = (const v16h*)(wkt + (size_t)nt * 512);
#pragma unroll
    for (int kt = 0; kt < KT_K; ++kt) {
      v16h a = ldsA_f16(valb, mt * 16, kt * 32, 256, lane);
      v16h b = bp[(size_t)kt * NT_K * 32 + lane];
      acc = wmma_f16(a, b, acc);
    }
    int n = nt * 16 + (lane & 15);
    int m0 = mt * 16 + ((lane >> 4) * 8);
    if (n < 200) {
      float w = qw[n], bb = bk[n];
#pragma unroll
      for (int r = 0; r < 8; ++r)
        atomicAdd(&sArr[m0 + r], w * tanhf(acc[r] + bb));   // ds_add_f32
    }
  }
  __syncthreads();
  if (tid == 0) {
    float mx = -1e30f;
    for (int l = 0; l < 32; ++l) mx = fmaxf(mx, scale * sArr[l]);
    float sum = 0.f;
    for (int l = 0; l < 32; ++l) sum += __expf(scale * sArr[l] - mx);
    red[0] = mx; red[1] = 1.f / sum;
  }
  __syncthreads();
  if (tid < 32) aArr[tid] = __expf(scale * sArr[tid] - red[0]) * red[1];
  __syncthreads();
  {
    float acc = 0.f;
#pragma unroll 4
    for (int l = 0; l < 32; ++l) acc += aArr[l] * (float)valb[l * 256 + tid];
    attnOut[(size_t)slot * 256 + tid] = acc;
  }
}

// ---------------- fusion: hard-gumbel select + cosine WMMA + KNRM -----------

__global__ void __launch_bounds__(256)
fuse_kernel(const int* __restrict__ cand, const int* __restrict__ clicked,
            const unsigned char* __restrict__ mask, const float* __restrict__ emb,
            const float* __restrict__ attn, const float* __restrict__ Wltr,
            const float* __restrict__ bltr, float* __restrict__ scores) {
  __shared__ __align__(32) _Float16 cddF[32 * DPAD];
  __shared__ __align__(32) _Float16 hisF[32 * DPAD];
  __shared__ float dotsL[32 * 32];
  __shared__ float cA[256], wL[50], na[32], nb[32], pooled[20];
  __shared__ int sel;

  const int tid = threadIdx.x, wave = tid >> 5, lane = tid & 31;
  const int bc = blockIdx.x;          // b*CDD + c
  const int b  = bc / 5;

  cA[tid] = attn[(size_t)bc * 256 + tid];
  if (tid < 20) pooled[tid] = 0.f;
  __syncthreads();

  // w[h] = cdd_attn . his_attn[h] (+ gumbel); forward pass of ST-gumbel = argmax
  for (int h = wave; h < 50; h += 8) {
    const float* ha = attn + (size_t)(NCAND + b * 50 + h) * 256;
    float d = 0.f;
    for (int v = lane; v < 256; v += 32) d += cA[v] * ha[v];
#pragma unroll
    for (int off = 16; off > 0; off >>= 1) d += __shfl_xor(d, off, 32);
    if (lane == 0) {
      if (mask[b * 50 + h]) wL[h] = -1e30f;
      else wL[h] = d + gumbel_hash((unsigned)(bc * 50 + h) * 2654435761u + 12345u);
    }
  }
  __syncthreads();
  if (tid == 0) {
    int best = 0; float bv = wL[0];
    for (int h = 1; h < 50; ++h) if (wL[h] > bv) { bv = wL[h]; best = h; }
    sel = best;
  }
  __syncthreads();

  const int* ct = cand + (size_t)bc * 32;
  const int* ht = clicked + (size_t)(b * 50 + sel) * 32;
  for (int idx = tid; idx < 32 * DPAD; idx += 256) {
    int r = idx / DPAD, d = idx - r * DPAD;
    cddF[idx] = (d < 300) ? (_Float16)emb[(size_t)ct[r] * 300 + d] : (_Float16)0.f;
    hisF[idx] = (d < 300) ? (_Float16)emb[(size_t)ht[r] * 300 + d] : (_Float16)0.f;
  }
  __syncthreads();

  // dots = cdd @ his^T  (32 x 32, K=320) via WMMA
  if (wave < 4) {
    int mt = wave >> 1, nt = wave & 1;
    v8f acc = {};
#pragma unroll
    for (int kt = 0; kt < KT_Q; ++kt) {
      v16h a  = ldsA_f16(cddF, mt * 16, kt * 32, DPAD, lane);
      v16h b2 = ldsB_rows(hisF, kt * 32, nt * 16, DPAD, lane);
      acc = wmma_f16(a, b2, acc);
    }
    stC_f32(dotsL, acc, mt * 16, nt * 16, 32, lane, 1.0f);
  }
  if (tid < 64) {
    int i = tid & 31;
    const _Float16* row = (tid < 32) ? (cddF + i * DPAD) : (hisF + i * DPAD);
    float s = 0.f;
    for (int d = 0; d < 300; ++d) { float x = (float)row[d]; s += x * x; }
    if (tid < 32) na[i] = sqrtf(s); else nb[i] = sqrtf(s);
  }
  __syncthreads();

  // KNRM: 20 RBF kernels, sum_j -> log -> sum_i
  if (tid < 32) {
    float ks[20];
#pragma unroll
    for (int k = 0; k < 20; ++k) ks[k] = 0.f;
    float nai = na[tid];
    for (int j = 0; j < 32; ++j) {
      float m = dotsL[tid * 32 + j] / fmaxf(nai * nb[j], 1e-8f);
#pragma unroll
      for (int k = 0; k < 20; ++k) {
        float mu   = 0.1f * (float)k - 0.9f;
        float inv2 = (k == 19) ? 500000.0f : 50.0f;   // 1/(2*sigma^2)
        float df   = m - mu;
        ks[k] += __expf(-df * df * inv2);
      }
    }
#pragma unroll
    for (int k = 0; k < 20; ++k)
      atomicAdd(&pooled[k], __logf(fmaxf(ks[k], 1e-10f)));
  }
  __syncthreads();
  if (tid == 0) {
    float sc = bltr[0];
    for (int k = 0; k < 20; ++k) sc += pooled[k] * Wltr[k];
    scores[bc] = sc;
  }
}

// ---------------- final log-softmax over CDD --------------------------------

__global__ void lsm_kernel(const float* __restrict__ scores,
                           float* __restrict__ out) {
  int b = threadIdx.x;
  if (b >= 32) return;
  float v[5], mx = -1e30f;
  for (int c = 0; c < 5; ++c) { v[c] = scores[b * 5 + c]; mx = fmaxf(mx, v[c]); }
  float sum = 0.f;
  for (int c = 0; c < 5; ++c) sum += __expf(v[c] - mx);
  float lse = mx + __logf(sum);
  for (int c = 0; c < 5; ++c) out[b * 5 + c] = v[c] - lse;
}

// ---------------- launch ----------------------------------------------------

extern "C" void kernel_launch(void* const* d_in, const int* in_sizes, int n_in,
                              void* d_out, int out_size, void* d_ws,
                              size_t ws_size, hipStream_t stream) {
  const int*           cand    = (const int*)d_in[0];
  const int*           clicked = (const int*)d_in[1];
  const unsigned char* mask    = (const unsigned char*)d_in[2];
  const float*         emb     = (const float*)d_in[3];
  const float*         qwv     = (const float*)d_in[4];
  const float*         Wq      = (const float*)d_in[5];
  const float*         Wv      = (const float*)d_in[6];
  const float*         Wk      = (const float*)d_in[7];
  const float*         bk      = (const float*)d_in[8];
  const float*         Wltr    = (const float*)d_in[9];
  const float*         bltr    = (const float*)d_in[10];

  constexpr size_t WQT_HALFS = (size_t)16 * KT_Q * NT_Q * 512;  // 1,638,400
  constexpr size_t WVT_HALFS = (size_t)16 * KT_Q * 512;         //    81,920
  constexpr size_t WKT_HALFS = (size_t)KT_K * NT_K * 512;       //    53,248

  _Float16* wqt = (_Float16*)d_ws;
  _Float16* wvt = wqt + WQT_HALFS;
  _Float16* wkt = wvt + WVT_HALFS;
  float* attn   = (float*)(wkt + WKT_HALFS);          // 1760*256 floats
  float* scores = attn + (size_t)NITEM * 256;         // 160 floats

  prep_wq<<<(int)((WQT_HALFS + 255) / 256), 256, 0, stream>>>(Wq, wqt);
  prep_wv<<<(int)((WVT_HALFS + 255) / 256), 256, 0, stream>>>(Wv, wvt);
  prep_wk<<<(int)((WKT_HALFS + 255) / 256), 256, 0, stream>>>(Wk, wkt);

  encode_kernel<<<NITEM, 256, 0, stream>>>(cand, clicked, emb, qwv, bk,
                                           wqt, wvt, wkt, attn);
  fuse_kernel<<<NCAND, 256, 0, stream>>>(cand, clicked, mask, emb, attn,
                                         Wltr, bltr, scores);
  lsm_kernel<<<1, 32, 0, stream>>>(scores, (float*)d_out);
}
